// RNNModel_42614665511221
// MI455X (gfx1250) — matrix operand order repaired and checked by reading, not compiled, not run on hardware
//
#include <hip/hip_runtime.h>
#include <hip/hip_bf16.h>

// ---------------------------------------------------------------------------
// Types for CDNA5 WMMA (wave32): D(16x16 f32) = A(16x32 bf16) * B(32x16 bf16) + C
// ---------------------------------------------------------------------------
typedef __attribute__((ext_vector_type(16))) __bf16 bf16x16;
typedef __attribute__((ext_vector_type(8)))  float  f32x8;

union FragB {
  bf16x16 v;
  uint4   q[2];
};

// Problem constants
#define SEQT   70
#define BATCH  16
#define NINP   400
#define NHID   1150
#define NTOK   33278
#define MROWS  (SEQT * BATCH)        // 1120
#define KINP_P 416                   // 400 padded to x32
#define KHID_P 1152                  // 1150 padded to x32
#define NTOK_P 33280                 // 33278 padded to x16
#define C0_CONST 367.5754132818691f  // NINP/2 * log(2*pi)

// ---------------------------------------------------------------------------
// WMMA cores. One wave computes a 16x64 (core) or 32x64 (core2) strip of
// C = A(M,K) @ B(N,K)^T.
// A-fragment (16-bit, ISA 7.12.2): lane row = lane&15,
//   V0..3 hold K = 8*(lane>>4)..+7, V4..7 hold K = 16 + 8*(lane>>4)..+7
// B-fragment: lane col = lane&15, same K layout as the A-fragment
//   (V0..3 K = 8*(lane>>4)..+7, V4..7 K = 16 + 8*(lane>>4)..+7).
// ---------------------------------------------------------------------------
__device__ __forceinline__ void wmma_core(const __bf16* __restrict__ A, int lda,
                                          const __bf16* __restrict__ B, int ldb,
                                          int mtile, int ng, int ksteps,
                                          f32x8* acc) {
  const int lane = threadIdx.x & 31;
  const int lr = lane & 15;
  const int kh = (lane >> 4) << 3;
  const int kb = (lane >> 4) << 4;
  const __bf16* ap  = A + (size_t)(mtile * 16 + lr) * lda + kh;
  const __bf16* bp0 = B + (size_t)(ng * 64 + lr) * ldb + kh;
  const size_t bstep = (size_t)16 * ldb;
  for (int kk = 0; kk < ksteps; ++kk) {
    FragB fa;
    fa.q[0] = *(const uint4*)(ap);
    fa.q[1] = *(const uint4*)(ap + 16);
    ap += 32;
    const __bf16* bp = bp0;
#pragma unroll
    for (int s = 0; s < 4; ++s) {
      FragB fb;
      fb.q[0] = *(const uint4*)(bp);
      fb.q[1] = *(const uint4*)(bp + 16);
      acc[s] = __builtin_amdgcn_wmma_f32_16x16x32_bf16(
          false, fa.v, false, fb.v, (short)0, acc[s], false, false);
      bp += bstep;
    }
    bp0 += 32;
  }
}

// dual-M variant: every B fragment feeds two WMMAs (halves B load traffic)
__device__ __forceinline__ void wmma_core2(const __bf16* __restrict__ A, int lda,
                                           const __bf16* __restrict__ B, int ldb,
                                           int mtile0, int ng, int ksteps,
                                           f32x8* acc0, f32x8* acc1) {
  const int lane = threadIdx.x & 31;
  const int lr = lane & 15;
  const int kh = (lane >> 4) << 3;
  const int kb = (lane >> 4) << 4;
  const __bf16* ap0 = A + (size_t)(mtile0 * 16 + lr) * lda + kh;
  const __bf16* ap1 = ap0 + (size_t)16 * lda;
  const __bf16* bp0 = B + (size_t)(ng * 64 + lr) * ldb + kh;
  const size_t bstep = (size_t)16 * ldb;
  for (int kk = 0; kk < ksteps; ++kk) {
    FragB fa0, fa1;
    fa0.q[0] = *(const uint4*)(ap0);
    fa0.q[1] = *(const uint4*)(ap0 + 16);
    ap0 += 32;
    fa1.q[0] = *(const uint4*)(ap1);
    fa1.q[1] = *(const uint4*)(ap1 + 16);
    ap1 += 32;
    const __bf16* bp = bp0;
#pragma unroll
    for (int s = 0; s < 4; ++s) {
      FragB fb;
      fb.q[0] = *(const uint4*)(bp);
      fb.q[1] = *(const uint4*)(bp + 16);
      acc0[s] = __builtin_amdgcn_wmma_f32_16x16x32_bf16(
          false, fa0.v, false, fb.v, (short)0, acc0[s], false, false);
      acc1[s] = __builtin_amdgcn_wmma_f32_16x16x32_bf16(
          false, fa1.v, false, fb.v, (short)0, acc1[s], false, false);
      bp += bstep;
    }
    bp0 += 32;
  }
}

// ---------------------------------------------------------------------------
// Utility kernels
// ---------------------------------------------------------------------------
__global__ void k_zero(unsigned int* p, long long n) {
  long long i = (long long)blockIdx.x * blockDim.x + threadIdx.x;
  if (i < n) p[i] = 0u;
}

__global__ void k_copyf(float* dst, const float* src, long long n) {
  long long i = (long long)blockIdx.x * blockDim.x + threadIdx.x;
  if (i < n) dst[i] = src[i];
}

// f32 -> bf16 with row/col zero padding
__global__ void k_cvt(const float* __restrict__ src, int srcStride, int srcOff,
                      int rows, int K, __bf16* __restrict__ dst,
                      long long total, int Kp) {
  long long i = (long long)blockIdx.x * blockDim.x + threadIdx.x;
  if (i >= total) return;
  int r = (int)(i / Kp), k = (int)(i % Kp);
  float v = (r < rows && k < K) ? src[(size_t)r * srcStride + srcOff + k] : 0.f;
  dst[i] = (__bf16)v;
}

__global__ void k_embed(const int* __restrict__ tok, const float* __restrict__ emb,
                        __bf16* __restrict__ dst, long long total) {
  long long i = (long long)blockIdx.x * blockDim.x + threadIdx.x;
  if (i >= total) return;
  int m = (int)(i / KINP_P), k = (int)(i % KINP_P);
  float v = (k < NINP) ? emb[(size_t)tok[m] * NINP + k] : 0.f;
  dst[i] = (__bf16)v;
}

__global__ void k_rowsq(const float* __restrict__ src, int rows, int K, float* out) {
  int r = blockIdx.x * blockDim.x + threadIdx.x;
  if (r >= rows) return;
  const float* p = src + (size_t)r * K;
  float s = 0.f;
  for (int k = 0; k < K; ++k) { float v = p[k]; s += v * v; }
  out[r] = s;
}

__global__ void k_prep_cnf(const float* __restrict__ cnfW, float* wt, float* dwx) {
  int n = blockIdx.x * blockDim.x + threadIdx.x;
  if (n >= 448) return;
  wt[n]  = (n < NINP) ? cnfW[(size_t)n * (NINP + 1)] : 0.f;
  dwx[n] = (n < NINP) ? cnfW[(size_t)n * (NINP + 1) + 1 + n] : 0.f;
}

// initial bf16 staging of z (pads written as zero)
__global__ void k_stage(const float* __restrict__ z, __bf16* __restrict__ dst,
                        long long total) {
  long long i = (long long)blockIdx.x * blockDim.x + threadIdx.x;
  if (i >= total) return;
  int m = (int)(i / KINP_P), k = (int)(i % KINP_P);
  float v = (m < NTOK && k < NINP) ? z[(size_t)m * NINP + k] : 0.f;
  dst[i] = (__bf16)v;
}

// dl += w*div ; div = 0
__global__ void k_fin(float* __restrict__ dl, float* __restrict__ divb, float w, int n) {
  int i = blockIdx.x * blockDim.x + threadIdx.x;
  if (i >= n) return;
  dl[i] += w * divb[i];
  divb[i] = 0.f;
}

// ---------------------------------------------------------------------------
// GEMM kernels (blockDim 256; 8 waves/block; one wave -> 32x64 output strip)
// ---------------------------------------------------------------------------

// C = A@B^T + (b1[n] + b2[n])      (LSTM input projection, all timesteps)
__global__ void k_gemm_bias(const __bf16* __restrict__ A, int lda,
                            const __bf16* __restrict__ B, int ldb,
                            int ksteps, int ngroups, int nwaves,
                            const float* __restrict__ b1, const float* __restrict__ b2,
                            float* __restrict__ C, int ldc, int Nreal) {
  int wid = (blockIdx.x * blockDim.x + threadIdx.x) >> 5;
  if (wid >= nwaves) return;
  int mpair = wid / ngroups, ng = wid % ngroups;
  f32x8 acc[2][4] = {};
  wmma_core2(A, lda, B, ldb, mpair * 2, ng, ksteps, acc[0], acc[1]);
  const int lane = threadIdx.x & 31;
  const int nl = lane & 15, mh = (lane >> 4) << 3;
#pragma unroll
  for (int hh = 0; hh < 2; ++hh) {
    int mbase = (mpair * 2 + hh) * 16 + mh;
#pragma unroll
    for (int s = 0; s < 4; ++s) {
      int n = ng * 64 + s * 16 + nl;
      if (n >= Nreal) continue;
      float bias = b1[n] + b2[n];
#pragma unroll
      for (int v = 0; v < 8; ++v)
        C[(size_t)(mbase + v) * ldc + n] = acc[hh][s][v] + bias;
    }
  }
}

// out = dot - 0.5*(e2[n] + h2n[m]) - C0 - dl[n]     (pre-softmax logits)
__global__ void k_gemm_logits(const __bf16* __restrict__ A, int lda,
                              const __bf16* __restrict__ B, int ldb,
                              int ksteps, int ngroups, int nwaves,
                              const float* __restrict__ e2, const float* __restrict__ h2n,
                              const float* __restrict__ dl, float* __restrict__ out) {
  int wid = (blockIdx.x * blockDim.x + threadIdx.x) >> 5;
  if (wid >= nwaves) return;
  int mpair = wid / ngroups, ng = wid % ngroups;
  f32x8 acc[2][4] = {};
  wmma_core2(A, lda, B, ldb, mpair * 2, ng, ksteps, acc[0], acc[1]);
  const int lane = threadIdx.x & 31;
  const int nl = lane & 15, mh = (lane >> 4) << 3;
#pragma unroll
  for (int hh = 0; hh < 2; ++hh) {
    int mbase = (mpair * 2 + hh) * 16 + mh;
#pragma unroll
    for (int s = 0; s < 4; ++s) {
      int n = ng * 64 + s * 16 + nl;
      if (n >= NTOK) continue;
      float cn = -0.5f * e2[n] - C0_CONST - dl[n];
#pragma unroll
      for (int v = 0; v < 8; ++v) {
        int m = mbase + v;
        out[(size_t)m * NTOK + n] = acc[hh][s][v] + cn - 0.5f * h2n[m];
      }
    }
  }
}

// CNF eval with fully fused RK4 plumbing:
//   pre = z_eval@Wx^T + t*wt + b ; k = relu(pre)
//   div[m] += sum_n (pre>0)*dWx[n]           (wave-reduced, then atomics)
//   zn += cacc*k                             (RK4 accumulator)
//   if last: zc = zn ; stage bf16(zn)        (commit z', stage next step's eval1)
//   else:    stage bf16(zc + cstage*k)       (stage next eval's input)
__global__ void k_gemm_cnf(const __bf16* __restrict__ A, int lda,
                           const __bf16* __restrict__ B, int ldb,
                           int ksteps, int ngroups, int nwaves,
                           float tval, float cstage, float cacc, int last,
                           const float* __restrict__ wt, const float* __restrict__ dwx,
                           const float* __restrict__ bias,
                           float* __restrict__ zc, float* __restrict__ zn,
                           __bf16* __restrict__ zbf_out, float* __restrict__ divb) {
  int wid = (blockIdx.x * blockDim.x + threadIdx.x) >> 5;
  if (wid >= nwaves) return;
  int mpair = wid / ngroups, ng = wid % ngroups;
  f32x8 acc[2][4] = {};
  wmma_core2(A, lda, B, ldb, mpair * 2, ng, ksteps, acc[0], acc[1]);
  const int lane = threadIdx.x & 31;
  const int nl = lane & 15, mh = (lane >> 4) << 3;
#pragma unroll
  for (int hh = 0; hh < 2; ++hh) {
    int mbase = (mpair * 2 + hh) * 16 + mh;
    float divacc[8];
#pragma unroll
    for (int v = 0; v < 8; ++v) divacc[v] = 0.f;
#pragma unroll
    for (int s = 0; s < 4; ++s) {
      int n = ng * 64 + s * 16 + nl;
      bool nok = n < NINP;
      float wtn = nok ? wt[n] : 0.f;
      float bn  = nok ? bias[n] : 0.f;
      float dwn = nok ? dwx[n] : 0.f;
#pragma unroll
      for (int v = 0; v < 8; ++v) {
        float pre = acc[hh][s][v] + tval * wtn + bn;
        float kv = pre > 0.f ? pre : 0.f;
        if (nok) {
          size_t o = (size_t)(mbase + v) * NINP + n;
          float znv = zn[o] + cacc * kv;
          zn[o] = znv;
          float sv;
          if (last) { zc[o] = znv; sv = znv; }
          else      { sv = zc[o] + cstage * kv; }
          zbf_out[(size_t)(mbase + v) * KINP_P + n] = (__bf16)sv;
          if (pre > 0.f) divacc[v] += dwn;
        }
      }
    }
#pragma unroll
    for (int v = 0; v < 8; ++v) {       // reduce over 16 columns (half-wave)
      float d = divacc[v];
      d += __shfl_xor(d, 1, 32);
      d += __shfl_xor(d, 2, 32);
      d += __shfl_xor(d, 4, 32);
      d += __shfl_xor(d, 8, 32);
      if (nl == 0) atomicAdd(divb + mbase + v, d);
    }
  }
}

// ---------------------------------------------------------------------------
// Persistent LSTM layer kernel: runs all 70 timesteps in one launch.
// Phase A (gates): wave w adds h@Whh^T into XW rows of timestep t.
// Phase B (cell):  fused gate nonlinearities + state update + bf16 staging.
// Inter-block sync: fenced monotonic atomic counter barrier (blocks co-resident).
// ---------------------------------------------------------------------------
__device__ __forceinline__ void gbar(unsigned* bar, unsigned nblocks, unsigned* phase) {
  __threadfence();
  __syncthreads();
  if (threadIdx.x == 0) {
    atomicAdd(bar, 1u);
    unsigned target = (++(*phase)) * nblocks;
    while (atomicAdd(bar, 0u) < target) __builtin_amdgcn_s_sleep(1);
  }
  __syncthreads();
  __threadfence();
}

__device__ __forceinline__ float sigf(float x) { return 1.f / (1.f + expf(-x)); }

__global__ void __launch_bounds__(256)
k_lstm_layer(const __bf16* __restrict__ WhhBf, int Hp, int ksteps,
             int H, int Nreal, int ldc, int ngroups,
             float* __restrict__ XW,
             const float* __restrict__ h0, const float* __restrict__ c0,
             float* __restrict__ hw, float* __restrict__ cw,
             __bf16* __restrict__ hbf,
             __bf16* __restrict__ ybf, float* __restrict__ yf,
             unsigned* bar, int nblocks) {
  const int tid = threadIdx.x;
  const int gtid = blockIdx.x * 256 + tid;
  const int gthreads = nblocks * 256;
  unsigned phase = 0;

  // init state: h/c f32 copies + bf16 h staging (pads zeroed)
  for (int i = gtid; i < 16 * Hp; i += gthreads) {
    int b = i / Hp, j = i % Hp;
    hbf[i] = (__bf16)((j < H) ? h0[b * H + j] : 0.f);
  }
  for (int i = gtid; i < 16 * H; i += gthreads) {
    hw[i] = h0[i];
    cw[i] = c0[i];
  }
  gbar(bar, nblocks, &phase);

  const int lane = tid & 31;
  const int nl = lane & 15, mh = (lane >> 4) << 3;
  const int w = gtid >> 5;

  for (int t = 0; t < SEQT; ++t) {
    // -------- phase A: gates += h @ Whh^T (M = 16 batch rows) --------
    if (w < ngroups) {
      f32x8 acc[4] = {};
      wmma_core(hbf, Hp, WhhBf, Hp, 0, w, ksteps, acc);
#pragma unroll
      for (int s = 0; s < 4; ++s) {
        int n = w * 64 + s * 16 + nl;
        if (n >= Nreal) continue;
#pragma unroll
        for (int v = 0; v < 8; ++v) {
          size_t o = (size_t)(t * 16 + v + mh) * ldc + n;
          XW[o] = XW[o] + acc[s][v];
        }
      }
    }
    gbar(bar, nblocks, &phase);

    // -------- phase B: cell update --------
    for (int i = gtid; i < 16 * H; i += gthreads) {
      int b = i / H, j = i % H;
      const float* gr = XW + (size_t)(t * 16 + b) * ldc;
      float gi = gr[j], gf = gr[H + j], gg = gr[2 * H + j], go = gr[3 * H + j];
      float cn = sigf(gf) * cw[i] + sigf(gi) * tanhf(gg);
      float hn = sigf(go) * tanhf(cn);
      cw[i] = cn;
      hw[i] = hn;
      hbf[(size_t)b * Hp + j] = (__bf16)hn;
      ybf[(size_t)(t * 16 + b) * Hp + j] = (__bf16)hn;
      if (yf) yf[(size_t)(t * 16 + b) * H + j] = hn;
    }
    gbar(bar, nblocks, &phase);
  }
}

// ---------------------------------------------------------------------------
// Row softmax + log(p + 1e-8), in place. One block (256 thr) per row.
// ---------------------------------------------------------------------------
__global__ void k_softmax_log(float* __restrict__ out, int N) {
  __shared__ float red[8];
  float* p = out + (size_t)blockIdx.x * N;
  int tid = threadIdx.x;
  float mx = -3.4e38f;
  for (int i = tid; i < N; i += 256) mx = fmaxf(mx, p[i]);
#pragma unroll
  for (int m = 16; m >= 1; m >>= 1) mx = fmaxf(mx, __shfl_xor(mx, m, 32));
  if ((tid & 31) == 0) red[tid >> 5] = mx;
  __syncthreads();
  if (tid == 0) {
    float v = red[0];
    for (int i = 1; i < 8; ++i) v = fmaxf(v, red[i]);
    red[0] = v;
  }
  __syncthreads();
  mx = red[0];
  __syncthreads();
  float sum = 0.f;
  for (int i = tid; i < N; i += 256) sum += expf(p[i] - mx);
#pragma unroll
  for (int m = 16; m >= 1; m >>= 1) sum += __shfl_xor(sum, m, 32);
  if ((tid & 31) == 0) red[tid >> 5] = sum;
  __syncthreads();
  if (tid == 0) {
    float v = 0.f;
    for (int i = 0; i < 8; ++i) v += red[i];
    red[0] = v;
  }
  __syncthreads();
  float lz = logf(red[0]);
  for (int i = tid; i < N; i += 256) {
    float pr = expf(p[i] - mx - lz);
    p[i] = logf(pr + 1e-8f);
  }
}

// ---------------------------------------------------------------------------
// Host orchestration
// ---------------------------------------------------------------------------
#define L1D(kern, n, ...) \
  kern<<<dim3((unsigned)((((long long)(n)) + 255) / 256)), dim3(256), 0, stream>>>(__VA_ARGS__)

extern "C" void kernel_launch(void* const* d_in, const int* in_sizes, int n_in,
                              void* d_out, int out_size, void* d_ws, size_t ws_size,
                              hipStream_t stream) {
  (void)in_sizes; (void)n_in; (void)out_size; (void)ws_size;
  const int*   tok  = (const int*)  d_in[0];
  const float* h0_0 = (const float*)d_in[1];
  const float* c0_0 = (const float*)d_in[2];
  const float* h0_1 = (const float*)d_in[3];
  const float* c0_1 = (const float*)d_in[4];
  const float* h0_2 = (const float*)d_in[5];
  const float* c0_2 = (const float*)d_in[6];
  const float* embW = (const float*)d_in[7];
  const float* Wih0 = (const float*)d_in[8];
  const float* Whh0 = (const float*)d_in[9];
  const float* bih0 = (const float*)d_in[10];
  const float* bhh0 = (const float*)d_in[11];
  const float* Wih1 = (const float*)d_in[12];
  const float* Whh1 = (const float*)d_in[13];
  const float* bih1 = (const float*)d_in[14];
  const float* bhh1 = (const float*)d_in[15];
  const float* Wih2 = (const float*)d_in[16];
  const float* Whh2 = (const float*)d_in[17];
  const float* bih2 = (const float*)d_in[18];
  const float* bhh2 = (const float*)d_in[19];
  const float* cnfW = (const float*)d_in[20];
  const float* cnfb = (const float*)d_in[21];

  char* base = (char*)d_ws;
  size_t off = 0;
  auto alloc = [&](size_t bytes) -> void* {
    void* p = base + off;
    off = (off + bytes + 255) & ~(size_t)255;
    return p;
  };
  // bf16 staging buffers
  __bf16* x0bf   = (__bf16*)alloc((size_t)MROWS * KINP_P * 2);
  __bf16* y0bf   = (__bf16*)alloc((size_t)MROWS * KHID_P * 2);
  __bf16* y1bf   = (__bf16*)alloc((size_t)MROWS * KHID_P * 2);
  __bf16* y2bf   = (__bf16*)alloc((size_t)MROWS * KINP_P * 2);
  __bf16* hbf0   = (__bf16*)alloc((size_t)16 * KHID_P * 2);
  __bf16* hbf1   = (__bf16*)alloc((size_t)16 * KHID_P * 2);
  __bf16* hbf2   = (__bf16*)alloc((size_t)16 * KINP_P * 2);
  __bf16* Wih0bf = (__bf16*)alloc((size_t)4608 * KINP_P * 2);
  __bf16* Whh0bf = (__bf16*)alloc((size_t)4608 * KHID_P * 2);
  __bf16* Wih1bf = (__bf16*)alloc((size_t)4608 * KHID_P * 2);
  __bf16* Whh1bf = (__bf16*)alloc((size_t)4608 * KHID_P * 2);
  __bf16* Wih2bf = (__bf16*)alloc((size_t)1600 * KHID_P * 2);
  __bf16* Whh2bf = (__bf16*)alloc((size_t)1600 * KINP_P * 2);
  __bf16* embbf  = (__bf16*)alloc((size_t)NTOK_P * KINP_P * 2);
  __bf16* Wxbf   = (__bf16*)alloc((size_t)448 * KINP_P * 2);
  __bf16* zbfA   = (__bf16*)alloc((size_t)NTOK_P * KINP_P * 2);
  __bf16* zbfB   = (__bf16*)alloc((size_t)NTOK_P * KINP_P * 2);
  // f32 buffers
  float* XW   = (float*)alloc((size_t)MROWS * 4608 * 4);
  float* hw0  = (float*)alloc((size_t)16 * NHID * 4);
  float* cw0  = (float*)alloc((size_t)16 * NHID * 4);
  float* hw1  = (float*)alloc((size_t)16 * NHID * 4);
  float* cw1  = (float*)alloc((size_t)16 * NHID * 4);
  float* hw2  = (float*)alloc((size_t)16 * NINP * 4);
  float* cw2  = (float*)alloc((size_t)16 * NINP * 4);
  float* y2f  = (float*)alloc((size_t)MROWS * NINP * 4);
  float* divb = (float*)alloc((size_t)NTOK_P * 4);
  float* dl   = (float*)alloc((size_t)NTOK_P * 4);
  float* e2   = (float*)alloc((size_t)NTOK_P * 4);
  float* h2n  = (float*)alloc((size_t)MROWS * 4);
  float* wtb  = (float*)alloc((size_t)448 * 4);
  float* dwxb = (float*)alloc((size_t)448 * 4);
  unsigned* bars = (unsigned*)alloc(4 * sizeof(unsigned));

  // ---- zero pad-sensitive buffers / counters (ws arrives poisoned) ----
  L1D(k_zero, (long long)MROWS * KHID_P / 2, (unsigned int*)y0bf, (long long)MROWS * KHID_P / 2);
  L1D(k_zero, (long long)MROWS * KHID_P / 2, (unsigned int*)y1bf, (long long)MROWS * KHID_P / 2);
  L1D(k_zero, (long long)MROWS * KINP_P / 2, (unsigned int*)y2bf, (long long)MROWS * KINP_P / 2);
  L1D(k_zero, (long long)NTOK_P * KINP_P / 2, (unsigned int*)zbfB, (long long)NTOK_P * KINP_P / 2);
  L1D(k_zero, NTOK_P, (unsigned int*)divb, NTOK_P);
  L1D(k_zero, NTOK_P, (unsigned int*)dl, NTOK_P);
  L1D(k_zero, 4, (unsigned int*)bars, 4);

  // ---- embedding + weight staging to bf16 ----
  L1D(k_embed, (long long)MROWS * KINP_P, tok, embW, x0bf, (long long)MROWS * KINP_P);
  L1D(k_cvt, (long long)4608 * KINP_P, Wih0, NINP, 0, 4600, NINP, Wih0bf, (long long)4608 * KINP_P, KINP_P);
  L1D(k_cvt, (long long)4608 * KHID_P, Whh0, NHID, 0, 4600, NHID, Whh0bf, (long long)4608 * KHID_P, KHID_P);
  L1D(k_cvt, (long long)4608 * KHID_P, Wih1, NHID, 0, 4600, NHID, Wih1bf, (long long)4608 * KHID_P, KHID_P);
  L1D(k_cvt, (long long)4608 * KHID_P, Whh1, NHID, 0, 4600, NHID, Whh1bf, (long long)4608 * KHID_P, KHID_P);
  L1D(k_cvt, (long long)1600 * KHID_P, Wih2, NHID, 0, 1600, NHID, Wih2bf, (long long)1600 * KHID_P, KHID_P);
  L1D(k_cvt, (long long)1600 * KINP_P, Whh2, NINP, 0, 1600, NINP, Whh2bf, (long long)1600 * KINP_P, KINP_P);
  L1D(k_cvt, (long long)NTOK_P * KINP_P, embW, NINP, 0, NTOK, NINP, embbf, (long long)NTOK_P * KINP_P, KINP_P);
  L1D(k_cvt, (long long)448 * KINP_P, cnfW, NINP + 1, 1, NINP, NINP, Wxbf, (long long)448 * KINP_P, KINP_P);
  L1D(k_prep_cnf, 448, cnfW, wtb, dwxb);

  // ---- LSTM: XW precompute GEMM (35 M-pairs) + one persistent kernel/layer ----
  // layer 0
  L1D(k_gemm_bias, 35LL * 72 * 32, x0bf, KINP_P, Wih0bf, KINP_P, 13, 72,
      35 * 72, bih0, bhh0, XW, 4608, 4600);
  k_lstm_layer<<<dim3(9), dim3(256), 0, stream>>>(
      Whh0bf, KHID_P, 36, NHID, 4600, 4608, 72, XW, h0_0, c0_0,
      hw0, cw0, hbf0, y0bf, (float*)nullptr, bars + 0, 9);
  // layer 1
  L1D(k_gemm_bias, 35LL * 72 * 32, y0bf, KHID_P, Wih1bf, KHID_P, 36, 72,
      35 * 72, bih1, bhh1, XW, 4608, 4600);
  k_lstm_layer<<<dim3(9), dim3(256), 0, stream>>>(
      Whh1bf, KHID_P, 36, NHID, 4600, 4608, 72, XW, h0_1, c0_1,
      hw1, cw1, hbf1, y1bf, (float*)nullptr, bars + 1, 9);
  // layer 2
  L1D(k_gemm_bias, 35LL * 25 * 32, y1bf, KHID_P, Wih2bf, KHID_P, 36, 25,
      35 * 25, bih2, bhh2, XW, 1600, 1600);
  k_lstm_layer<<<dim3(4), dim3(256), 0, stream>>>(
      Whh2bf, KINP_P, 13, NINP, 1600, 1600, 25, XW, h0_2, c0_2,
      hw2, cw2, hbf2, y2bf, y2f, bars + 2, 4);

  // ---- norms for squared distance ----
  L1D(k_rowsq, MROWS, y2f, MROWS, NINP, h2n);
  L1D(k_rowsq, NTOK, embW, NTOK, NINP, e2);

  // ---- CNF delta log p: RK4, fully fused into the WMMA GEMM epilogue.
  //      z (zc) and RK4 accumulator (zn) live in d_out scratch. ----
  float* zc = (float*)d_out;
  float* zn = zc + (size_t)NTOK_P * NINP;
  const long long ZN = (long long)NTOK * NINP;
  L1D(k_copyf, ZN, zc, embW, ZN);
  L1D(k_copyf, ZN, zn, embW, ZN);
  const long long stageN = (long long)NTOK_P * KINP_P;
  L1D(k_stage, stageN, zc, zbfA, stageN);        // zbf = bf16(z0)
  const float DT = 1.f / 8.f;
  const long long cnfWaves = 1040LL * 7;         // 1040 M-pairs x 7 N-groups
  __bf16* zin = zbfA;
  __bf16* zout = zbfB;
  for (int s2 = 0; s2 < 8; ++s2) {
    float t0 = s2 * DT;
    const float tv[4] = {t0, t0 + DT * 0.5f, t0 + DT * 0.5f, t0 + DT};
    const float cs[4] = {DT * 0.5f, DT * 0.5f, DT, 0.f};   // stage coef for NEXT eval
    const float ca[4] = {DT / 6.f, DT / 3.f, DT / 3.f, DT / 6.f};
    const float wq[4] = {1.f, 2.f, 2.f, 1.f};
    for (int e = 0; e < 4; ++e) {
      L1D(k_gemm_cnf, cnfWaves * 32, zin, KINP_P, Wxbf, KINP_P, 13, 7, (int)cnfWaves,
          tv[e], cs[e], ca[e], (e == 3) ? 1 : 0,
          wtb, dwxb, cnfb, zc, zn, zout, divb);
      L1D(k_fin, NTOK_P, dl, divb, -DT / 6.f * wq[e], NTOK_P);
      __bf16* tmp = zin; zin = zout; zout = tmp;
    }
  }

  // ---- logits GEMM (overwrites CNF scratch region of d_out) + softmax/log ----
  const long long logitWaves = 35LL * 520;       // 35 M-pairs x 520 N-groups
  L1D(k_gemm_logits, logitWaves * 32, y2bf, KINP_P, embbf, KINP_P, 13, 520,
      (int)logitWaves, e2, h2n, dl, (float*)d_out);
  k_softmax_log<<<dim3(MROWS), dim3(256), 0, stream>>>((float*)d_out, NTOK);

  // ---- final hidden/cell states to output tail ----
  float* tail = (float*)d_out + (size_t)MROWS * NTOK;
  L1D(k_copyf, 16LL * NHID, tail, hw0, 16LL * NHID); tail += 16 * NHID;
  L1D(k_copyf, 16LL * NHID, tail, cw0, 16LL * NHID); tail += 16 * NHID;
  L1D(k_copyf, 16LL * NHID, tail, hw1, 16LL * NHID); tail += 16 * NHID;
  L1D(k_copyf, 16LL * NHID, tail, cw1, 16LL * NHID); tail += 16 * NHID;
  L1D(k_copyf, 16LL * NINP, tail, hw2, 16LL * NINP); tail += 16 * NINP;
  L1D(k_copyf, 16LL * NINP, tail, cw2, 16LL * NINP);
}
